// SACBlock_22737556865492
// MI455X (gfx1250) — matrix operand order repaired and checked by reading, not compiled, not run on hardware
//
#include <hip/hip_runtime.h>
#include <hip/hip_bf16.h>

// ---- types for WMMA (gfx1250, wave32) ----
typedef __bf16 bf16_t;
typedef __attribute__((ext_vector_type(16))) __bf16 bf16x16;
typedef __attribute__((ext_vector_type(8)))  float  f32x8;
typedef __attribute__((__vector_size__(4 * sizeof(int)))) int async_v4i;

#define BN_EPS 1e-5f

#if defined(__HIP_DEVICE_COMPILE__) && __has_builtin(__builtin_amdgcn_global_load_async_to_lds_b128)
#define USE_ASYNC_LDS 1
#else
#define USE_ASYNC_LDS 0
#endif

// ---- problem dims ----
constexpr int Nn = 4, Cc = 64, Hh = 64, Ww = 512;
constexpr int C9 = 576;          // 9*C
constexpr int KATT = 160;        // 147 padded to 5 k-blocks of 32
constexpr int TEW = 64;          // pixel tile (one row segment)
constexpr int TILES = Nn * Hh * (Ww / TEW);   // 2048

// ---- workspace layout (bytes, 256B aligned regions) ----
constexpr size_t OFF_APACK_ATT = 0;          // 36*5*512 bf16  = 184320 B
constexpr size_t OFF_ATT_BIAS  = 184320;     // 576 f32        = 2304 B
constexpr size_t OFF_APACK_W1  = 186624;     // 4*18*512 bf16  = 73728 B
constexpr size_t OFF_SHIFT1    = 260352;     // 64 f32
constexpr size_t OFF_APACK_W2  = 260608;     // 4*18*512 bf16  = 73728 B
constexpr size_t OFF_SHIFT2    = 334336;     // 64 f32
constexpr size_t OFF_H1        = 334592;     // 131072*64 bf16 = 16777216 B

// ---- kernel A LDS layout (bytes) ----
constexpr size_t SM_B   = 0;                       // [64][160] bf16 = 20480
constexpr size_t SM_NF  = 20480;                   // [64][576] bf16 = 73728
constexpr size_t SM_F   = 94208;                   // [3][64][66] f32 = 50688
constexpr size_t SM_TO  = 144896;                  // tabOff[576] i32 = 2304
constexpr size_t SM_TB  = 147200;                  // tabBias[576] f32 = 2304
constexpr size_t SMEM_A = 149504;

// K index inside a 32-wide k-block for A-fragment element (lane, i).
// Lanes 0-15 hold K={0..15}, lanes 16-31 hold K={16..31}.
__device__ __forceinline__ int a_frag_k(int half, int i) {
    return half * 16 + i;
}

union pack8 { bf16_t h[8]; uint4 q; };

// =====================================================================
// prep: fold BN into weights/shifts, pack A fragments (bf16) into ws
// =====================================================================
__global__ void __launch_bounds__(256) sac_prep(
    const float* __restrict__ w_att, const float* __restrict__ b_att,
    const float* __restrict__ ag, const float* __restrict__ ab,
    const float* __restrict__ am, const float* __restrict__ av,
    const float* __restrict__ w1, const float* __restrict__ b1,
    const float* __restrict__ g1, const float* __restrict__ be1,
    const float* __restrict__ m1, const float* __restrict__ v1,
    const float* __restrict__ w2, const float* __restrict__ b2,
    const float* __restrict__ g2, const float* __restrict__ be2,
    const float* __restrict__ m2, const float* __restrict__ v2,
    char* __restrict__ ws)
{
    bf16_t* packAtt = (bf16_t*)(ws + OFF_APACK_ATT);
    float*  biasAtt = (float*)(ws + OFF_ATT_BIAS);
    bf16_t* packW1  = (bf16_t*)(ws + OFF_APACK_W1);
    float*  sh1     = (float*)(ws + OFF_SHIFT1);
    bf16_t* packW2  = (bf16_t*)(ws + OFF_APACK_W2);
    float*  sh2     = (float*)(ws + OFF_SHIFT2);

    const int NA = 36 * 5 * 512;   // att A-pack elements
    const int NW = 4 * 18 * 512;   // w1 / w2 A-pack elements
    int idx = blockIdx.x * blockDim.x + threadIdx.x;

    if (idx < NA) {  // attention weights, K = c*49 + ki*7 + kj (pad 147->160)
        int i = idx & 15, lane = (idx >> 4) & 31, blk = idx >> 9;
        int kb = blk % 5, mb = blk / 5;
        int M = mb * 16 + (lane & 15), half = lane >> 4;
        int K = kb * 32 + a_frag_k(half, i);
        float val = 0.f;
        if (K < 147) {
            int c = K / 49, r = K % 49;
            float sc = ag[M] * rsqrtf(av[M] + BN_EPS);
            val = w_att[(M * 3 + c) * 49 + r] * sc;
        }
        packAtt[idx] = (bf16_t)val;
        return;
    }
    idx -= NA;
    if (idx < NW) {  // w1: K = unfold channel (c*9 + off), natural order
        int i = idx & 15, lane = (idx >> 4) & 31, blk = idx >> 9;
        int kb = blk % 18, mb = blk / 18;
        int M = mb * 16 + (lane & 15), half = lane >> 4;
        int K = kb * 32 + a_frag_k(half, i);
        float sc = g1[M] * rsqrtf(v1[M] + BN_EPS);
        packW1[idx] = (bf16_t)(w1[M * 576 + K] * sc);
        return;
    }
    idx -= NW;
    if (idx < NW) {  // w2: reordered K = off*64 + c  (off = ki*3+kj)
        int i = idx & 15, lane = (idx >> 4) & 31, blk = idx >> 9;
        int kb = blk % 18, mb = blk / 18;
        int M = mb * 16 + (lane & 15), half = lane >> 4;
        int K = kb * 32 + a_frag_k(half, i);
        int off = K >> 6, c = K & 63;
        float sc = g2[M] * rsqrtf(v2[M] + BN_EPS);
        packW2[idx] = (bf16_t)(w2[(M * 64 + c) * 9 + off] * sc);
        return;
    }
    idx -= NW;
    if (idx < 576) {
        float sc = ag[idx] * rsqrtf(av[idx] + BN_EPS);
        biasAtt[idx] = b_att[idx] * sc + ab[idx] - am[idx] * sc;
        return;
    }
    idx -= 576;
    if (idx < 64) {
        float sc = g1[idx] * rsqrtf(v1[idx] + BN_EPS);
        sh1[idx] = b1[idx] * sc + be1[idx] - m1[idx] * sc;
        return;
    }
    idx -= 64;
    if (idx < 64) {
        float sc = g2[idx] * rsqrtf(v2[idx] + BN_EPS);
        sh2[idx] = b2[idx] * sc + be2[idx] - m2[idx] * sc;
    }
}

// =====================================================================
// Kernel A: attention GEMM -> sigmoid gate * unfold(feature) (LDS) ->
//           1x1 reduce GEMM -> ReLU -> h1 (bf16, pixel-major) to ws
// =====================================================================
__global__ void __launch_bounds__(256) sac_gate_reduce(
    const float* __restrict__ new_xyz, const float* __restrict__ feature,
    char* __restrict__ ws)
{
    extern __shared__ char smem[];
    bf16_t* ldsB   = (bf16_t*)(smem + SM_B);   // [TEW][KATT] im2col(new_xyz)
    bf16_t* ldsNF  = (bf16_t*)(smem + SM_NF);  // [TEW][C9] gated unfold panel
    float*  ldsF   = (float*)(smem + SM_F);    // [3][64ch][66] feature halo
    int*    tabOff = (int*)(smem + SM_TO);     // per-ch LDS gather offset
    float*  tabBias= (float*)(smem + SM_TB);   // per-ch folded bias

    const bf16_t* packAtt = (const bf16_t*)(ws + OFF_APACK_ATT);
    const float*  biasAtt = (const float*)(ws + OFF_ATT_BIAS);
    const bf16_t* packW1  = (const bf16_t*)(ws + OFF_APACK_W1);
    const float*  sh1     = (const float*)(ws + OFF_SHIFT1);
    bf16_t*       h1      = (bf16_t*)(ws + OFF_H1);

    int tile = blockIdx.x;
    int rowg = tile >> 3;             // W/TEW == 8 tiles per row
    int w0   = (tile & 7) * TEW;
    int n = rowg / Hh, h = rowg % Hh;
    int tid = threadIdx.x;

    // ---- phase 0a: bf16 im2col(new_xyz) B-panel, K = c*49 + ki*7 + kj ----
    for (int e = tid; e < TEW * KATT; e += 256) {
        int t = e / KATT, k = e % KATT;
        float v = 0.f;
        if (k < 147) {
            int c = k / 49, r = k % 49, ki = r / 7, kj = r % 7;
            int hh = h + ki - 3, wwp = w0 + t + kj - 3;
            if (hh >= 0 && hh < Hh && wwp >= 0 && wwp < Ww)
                v = new_xyz[((n * 3 + c) * Hh + hh) * Ww + wwp];
        }
        ldsB[t * KATT + k] = (bf16_t)v;
    }
    // ---- phase 0b: feature halo rows -> LDS (f32), layout [ri][c][wi] ----
    for (int e = tid; e < 3 * 64 * 66; e += 256) {
        int ri = e / (64 * 66), rem = e % (64 * 66);
        int c = rem / 66, wi = rem % 66;
        int hh = h + ri - 1, wwp = w0 + wi - 1;
        float v = 0.f;
        if (hh >= 0 && hh < Hh && wwp >= 0 && wwp < Ww)
            v = feature[((n * Cc + c) * Hh + hh) * Ww + wwp];
        ldsF[e] = v;
    }
    // ---- phase 0c: per-channel gather offset + bias tables ----
    for (int ch = tid; ch < C9; ch += 256) {
        int c = ch / 9, koff = ch % 9;
        int dh = koff / 3, dw = koff % 3;
        tabOff[ch] = (dh * 64 + c) * 66 + dw;   // +t at use site
        tabBias[ch] = biasAtt[ch];
    }
    __syncthreads();

    int lane = tid & 31, wave = tid >> 5;
    int half = lane >> 4, col = lane & 15;

    // ---- phase 1: attention GEMM (per-wave mb, A hoisted), gate -> NF ----
    for (int mb = wave; mb < 36; mb += 8) {
        const bf16_t* ap = packAtt + ((size_t)(mb * 5) * 32 + lane) * 16;
        bf16x16 afr[5];
#pragma unroll
        for (int kb = 0; kb < 5; kb++)
            afr[kb] = *(const bf16x16*)(ap + kb * 512);
        int chbase = mb * 16 + half * 8;
#pragma unroll
        for (int nb = 0; nb < 4; nb++) {
            f32x8 acc = {};
            const bf16_t* bp = ldsB + (nb * 16 + col) * KATT + half * 16;
#pragma unroll
            for (int kb = 0; kb < 5; kb++) {
                bf16x16 b = *(const bf16x16*)(bp + kb * 32);
                acc = __builtin_amdgcn_wmma_f32_16x16x32_bf16(
                    false, afr[kb], false, b, (short)0, acc, false, false);
            }
            int t = nb * 16 + col;
            pack8 pk;
#pragma unroll
            for (int r = 0; r < 8; r++) {
                int ch = chbase + r;                   // 0..575
                float x = acc[r] + tabBias[ch];
                float att = __builtin_amdgcn_rcpf(1.f + __expf(-x));
                pk.h[r] = (bf16_t)(att * ldsF[tabOff[ch] + t]);
            }
            // 8 contiguous channels, 16B aligned -> one b128 LDS store
            *(uint4*)((char*)ldsNF + ((size_t)t * C9 + chbase) * 2) = pk.q;
        }
    }
    __syncthreads();

    // ---- phase 2: 1x1 reduce GEMM (4 mb x 4 nb, K=576) + ReLU -> h1 ----
    for (int tI = wave; tI < 16; tI += 8) {
        int mb = tI >> 2, nb = tI & 3;
        f32x8 acc = {};
        const bf16_t* ap = packW1 + ((size_t)(mb * 18) * 32 + lane) * 16;
        const bf16_t* bp = ldsNF + (nb * 16 + col) * C9 + half * 16;
#pragma unroll
        for (int kb = 0; kb < 18; kb++) {
            __builtin_prefetch(ap + (kb + 1) * 512, 0, 1);
            bf16x16 a = *(const bf16x16*)(ap + kb * 512);
            bf16x16 b = *(const bf16x16*)(bp + kb * 32);
            acc = __builtin_amdgcn_wmma_f32_16x16x32_bf16(
                false, a, false, b, (short)0, acc, false, false);
        }
        int t = nb * 16 + col;
        long pix = (long)tile * TEW + t;               // flat (n,h,w)
        int chbase = mb * 16 + half * 8;
        pack8 pk;
#pragma unroll
        for (int r = 0; r < 8; r++) {
            float v = acc[r] + sh1[chbase + r];
            pk.h[r] = (bf16_t)(v > 0.f ? v : 0.f);
        }
        // 8 contiguous channels, 16B aligned -> one b128 global store
        *(uint4*)(h1 + pix * 64 + chbase) = pk.q;
    }
}

// =====================================================================
// Kernel B: 3x3 conv as GEMM (K = off*64 + c) + ReLU + residual -> out
// =====================================================================
__global__ void __launch_bounds__(256) sac_conv3x3(
    const float* __restrict__ feature, const char* __restrict__ ws,
    float* __restrict__ out)
{
    extern __shared__ char smem[];
    bf16_t* ldsH = (bf16_t*)smem;   // [3 rows][66 cols][64 ch] halo panel

    const bf16_t* packW2 = (const bf16_t*)(ws + OFF_APACK_W2);
    const float*  sh2    = (const float*)(ws + OFF_SHIFT2);
    const bf16_t* h1     = (const bf16_t*)(ws + OFF_H1);

    int tile = blockIdx.x;
    int rowg = tile >> 3;
    int w0   = (tile & 7) * TEW;
    int n = rowg / Hh, h = rowg % Hh;
    int tid = threadIdx.x;

    // ---- stage h1 halo ----
#if USE_ASYNC_LDS
    // zero-fill (covers OOB halo), then async b128 DMA for valid entries
    for (int e = tid; e < (3 * 66 * 64) / 4; e += 256)
        ((unsigned long long*)ldsH)[e] = 0ull;
    __syncthreads();
    for (int e = tid; e < 3 * 66 * 8; e += 256) {   // 16B segments
        int ri = e / (66 * 8), rem = e % (66 * 8);
        int wi = rem >> 3, seg = rem & 7;
        int hh = h + ri - 1, wwp = w0 + wi - 1;
        if (hh >= 0 && hh < Hh && wwp >= 0 && wwp < Ww) {
            const bf16_t* g = h1 + (((long)(n * Hh + hh)) * Ww + wwp) * 64 + seg * 8;
            bf16_t* l = ldsH + (ri * 66 + wi) * 64 + seg * 8;
            __builtin_amdgcn_global_load_async_to_lds_b128(
                (async_v4i*)g, (async_v4i*)l, 0, 0);
        }
    }
#if __has_builtin(__builtin_amdgcn_s_wait_asynccnt)
    __builtin_amdgcn_s_wait_asynccnt(0);
#else
    asm volatile("s_wait_asynccnt 0" ::: "memory");
#endif
#else
    for (int e = tid; e < 3 * 66 * 64; e += 256) {
        int ri = e / (66 * 64), rem = e % (66 * 64);
        int wi = rem >> 6, c = rem & 63;
        int hh = h + ri - 1, wwp = w0 + wi - 1;
        bf16_t v = (bf16_t)0.f;
        if (hh >= 0 && hh < Hh && wwp >= 0 && wwp < Ww)
            v = h1[(((long)(n * Hh + hh)) * Ww + wwp) * 64 + c];
        ldsH[e] = v;
    }
#endif
    __syncthreads();

    int lane = tid & 31, wave = tid >> 5;
    int half = lane >> 4, col = lane & 15;

    for (int tI = wave; tI < 16; tI += 8) {
        int mb = tI >> 2, nb = tI & 3;
        f32x8 acc = {};
        const bf16_t* ap = packW2 + ((size_t)(mb * 18) * 32 + lane) * 16;
        int t = nb * 16 + col;
#pragma unroll
        for (int kb = 0; kb < 18; kb++) {
            int off = kb >> 1;                 // 0..8 = ki*3+kj
            int ki = off / 3, kj = off % 3;
            int cbase = (kb & 1) * 32 + half * 16;
            __builtin_prefetch(ap + (kb + 1) * 512, 0, 1);
            bf16x16 a = *(const bf16x16*)(ap + kb * 512);
            bf16x16 b = *(const bf16x16*)(ldsH + (ki * 66 + (t + kj)) * 64 + cbase);
            acc = __builtin_amdgcn_wmma_f32_16x16x32_bf16(
                false, a, false, b, (short)0, acc, false, false);
        }
#pragma unroll
        for (int r = 0; r < 8; r++) {
            int ch = mb * 16 + half * 8 + r;
            float v = acc[r] + sh2[ch];
            v = v > 0.f ? v : 0.f;
            long fi = ((long)(n * Cc + ch) * Hh + h) * Ww + (w0 + t);
            out[fi] = v + feature[fi];
        }
    }
}

// =====================================================================
extern "C" void kernel_launch(void* const* d_in, const int* in_sizes, int n_in,
                              void* d_out, int out_size, void* d_ws, size_t ws_size,
                              hipStream_t stream)
{
    (void)in_sizes; (void)n_in; (void)out_size; (void)ws_size;
    const float* new_xyz = (const float*)d_in[1];
    const float* feature = (const float*)d_in[2];
    const float* w_att = (const float*)d_in[3];
    const float* b_att = (const float*)d_in[4];
    const float* ag = (const float*)d_in[5];
    const float* ab = (const float*)d_in[6];
    const float* am = (const float*)d_in[7];
    const float* av = (const float*)d_in[8];
    const float* w1 = (const float*)d_in[9];
    const float* b1 = (const float*)d_in[10];
    const float* g1 = (const float*)d_in[11];
    const float* be1 = (const float*)d_in[12];
    const float* m1 = (const float*)d_in[13];
    const float* v1 = (const float*)d_in[14];
    const float* w2 = (const float*)d_in[15];
    const float* b2 = (const float*)d_in[16];
    const float* g2 = (const float*)d_in[17];
    const float* be2 = (const float*)d_in[18];
    const float* m2 = (const float*)d_in[19];
    const float* v2 = (const float*)d_in[20];
    char* ws = (char*)d_ws;
    float* outp = (float*)d_out;

    size_t smemB = (size_t)3 * 66 * 64 * 2;    // 25344 B
    (void)hipFuncSetAttribute((const void*)sac_gate_reduce,
                              hipFuncAttributeMaxDynamicSharedMemorySize,
                              (int)SMEM_A);

    int prepN = 36 * 5 * 512 + 2 * 4 * 18 * 512 + 576 + 64 + 64;
    sac_prep<<<(prepN + 255) / 256, 256, 0, stream>>>(
        w_att, b_att, ag, ab, am, av,
        w1, b1, g1, be1, m1, v1,
        w2, b2, g2, be2, m2, v2, ws);

    sac_gate_reduce<<<TILES, 256, SMEM_A, stream>>>(new_xyz, feature, ws);
    sac_conv3x3<<<TILES, 256, smemB, stream>>>(feature, ws, outp);
}
